// GridAttention_2542620639912
// MI455X (gfx1250) — matrix operand order repaired and checked by reading, not compiled, not run on hardware
//
#include <hip/hip_runtime.h>
#include <hip/hip_fp16.h>

typedef _Float16 f16;
typedef __attribute__((ext_vector_type(16))) _Float16 v16h;
typedef __attribute__((ext_vector_type(8)))  _Float16 v8h;
typedef __attribute__((ext_vector_type(8)))  float    v8f;
typedef __attribute__((ext_vector_type(4)))  float    v4f;

#define Bb 16
#define Hh 128
#define Ww 128
#define Cc 256
#define TPB 256

// Packed weight fragment counts (f16 halves) in workspace
#define WQKV_HALF_COUNT (48*8*32*16)   // 196608 halves
#define WOUT_HALF_COUNT (16*8*32*16)   //  65536 halves

// LDS layout (bytes); strides in f16 halves
#define XN_STRIDE 264
#define Q_STRIDE  40
#define XN_OFF 0
#define XN_BYTES (64*XN_STRIDE*2)           // 33792 (reused as attn-out)
#define Q_OFF   (XN_OFF + XN_BYTES)
#define Q_BYTES (8*64*Q_STRIDE*2)           // 40960
#define KF_OFF  (Q_OFF + Q_BYTES)
#define KF_BYTES (8*4*32*16*2)              // 32768
#define VF_OFF  (KF_OFF + KF_BYTES)
#define VF_BYTES (8*2*2*32*16*2)            // 32768
#define PA_OFF  (VF_OFF + VF_BYTES)
#define PA_BYTES (8*64*64*2)                // 65536
#define SMEM_BYTES (PA_OFF + PA_BYTES)      // 205824

__device__ __forceinline__ v8f wmma16(v16h a, v16h b, v8f c) {
  return __builtin_amdgcn_wmma_f32_16x16x32_f16(false, a, false, b,
                                                (short)0, c, false, false);
}

// A-fragment (16-bit, 16x32): lane half h holds K=16h..16h+15 (contiguous); two 16B LDS loads
__device__ __forceinline__ v16h load_a_lds(const f16* row_base, int klo) {
  v8h x0 = *(const v8h*)(row_base + 2 * klo);
  v8h x1 = *(const v8h*)(row_base + 2 * klo + 8);
  v16h a;
#pragma unroll
  for (int i = 0; i < 8; ++i) { a[i] = x0[i]; a[i + 8] = x1[i]; }
  return a;
}

// Repack fp32 weights into WMMA B-fragment order (f16):
// B[k][n] = W[n][k]; lane holds col n = nt*16+(lane&15), k = kstep*32 + (lane>>4)*16 + h
__global__ void pack_weights_kernel(const float* __restrict__ in_proj_w,
                                    const float* __restrict__ out_proj_w,
                                    f16* __restrict__ wqkv_frag,
                                    f16* __restrict__ wout_frag) {
  int idx = blockIdx.x * blockDim.x + threadIdx.x;
  if (idx < WQKV_HALF_COUNT) {
    int h = idx & 15, frag = idx >> 4;
    int lane = frag & 31, t = frag >> 5;
    int kstep = t & 7, nt = t >> 3;
    int n = nt * 16 + (lane & 15);
    int k = kstep * 32 + (lane >> 4) * 16 + h;
    wqkv_frag[idx] = (f16)in_proj_w[n * Cc + k];
  } else {
    int j = idx - WQKV_HALF_COUNT;
    if (j < WOUT_HALF_COUNT) {
      int h = j & 15, frag = j >> 4;
      int lane = frag & 31, t = frag >> 5;
      int kstep = t & 7, nt = t >> 3;
      int n = nt * 16 + (lane & 15);
      int k = kstep * 32 + (lane >> 4) * 16 + h;
      wout_frag[j] = (f16)out_proj_w[n * Cc + k];
    }
  }
}

// One region (REGION: 0=Q,1=K,2=V) of the QKV GEMM; region is compile-time so the
// scatter epilogue has no divergent branches.
template <int REGION>
__device__ __forceinline__ void qkv_region(const f16* __restrict__ xn,
                                           f16* __restrict__ qs,
                                           f16* __restrict__ kf,
                                           f16* __restrict__ vf,
                                           const f16* __restrict__ wqkv_frag,
                                           const float* __restrict__ in_proj_b,
                                           int wid, int lane, int lidlow, int hi,
                                           int klo) {
#pragma unroll 1
  for (int j = 0; j < 2; ++j) {
    const int ntg = REGION * 16 + wid * 2 + j;
    v8f acc[4] = {};
#pragma unroll 1
    for (int ks = 0; ks < 8; ++ks) {
      const v16h bfr =
          *(const v16h*)(wqkv_frag + (((size_t)ntg * 8 + ks) * 32 + lane) * 16);
#pragma unroll
      for (int mt = 0; mt < 4; ++mt) {
        const v16h afr =
            load_a_lds(xn + (mt * 16 + lidlow) * XN_STRIDE + ks * 32, klo);
        acc[mt] = wmma16(afr, bfr, acc[mt]);
      }
    }
    const int n = ntg * 16 + lidlow;  // global feature index
    const float bias = in_proj_b[n];
#pragma unroll
    for (int mt = 0; mt < 4; ++mt) {
#pragma unroll
      for (int r = 0; r < 8; ++r) {
        const int m = mt * 16 + r + 8 * hi;  // token row (C-layout)
        const float v = acc[mt][r] + bias;
        if (REGION == 0) {                   // Q: row-major per head, pre-scaled
          const int hd = n >> 5, d = n & 31;
          qs[(hd * 64 + m) * Q_STRIDE + d] = (f16)(v * 0.17677669529663687f);
        } else if (REGION == 1) {            // K: scatter into B-fragment order
          const int nk = n - 256, hd = nk >> 5, d = nk & 31;
          const int tt = m >> 4;
          const int bl = (d >> 4) * 16 + (m & 15);
          kf[((hd * 4 + tt) * 32 + bl) * 16 + (d & 15)] = (f16)v;
        } else {                             // V: scatter into B-fragment order
          const int nv = n - 512, hd = nv >> 5, d = nv & 31;
          const int dt = d >> 4, dlow = d & 15;
          const int ksg = m >> 5, kl = m & 31;
          const int bl = (kl >> 4) * 16 + dlow;
          vf[(((hd * 2 + dt) * 2 + ksg) * 32 + bl) * 16 + (kl & 15)] = (f16)v;
        }
      }
    }
  }
}

__global__ void __launch_bounds__(TPB, 1)
grid_attn_kernel(const float* __restrict__ x,
                 const float* __restrict__ ln_w, const float* __restrict__ ln_b,
                 const f16* __restrict__ wqkv_frag, const float* __restrict__ in_proj_b,
                 const f16* __restrict__ wout_frag, const float* __restrict__ out_proj_b,
                 const float* __restrict__ gamma,
                 float* __restrict__ out) {
  extern __shared__ char smem[];
  f16* xn = (f16*)(smem + XN_OFF);   // normalized input; later reused as attn-out
  f16* qs = (f16*)(smem + Q_OFF);
  f16* kf = (f16*)(smem + KF_OFF);
  f16* vf = (f16*)(smem + VF_OFF);
  f16* pa = (f16*)(smem + PA_OFF);

  const int tid = threadIdx.x;
  const int lane = tid & 31;
  // wave id is uniform within the wave: pin it to an SGPR so per-wave bases are scalar
  const int wid = __builtin_amdgcn_readfirstlane(tid >> 5);
  const int lidlow = lane & 15;
  const int hi = lane >> 4;
  const int klo = hi ? 8 : 0;

  const int blk = blockIdx.x;          // 4096 windows
  const int wx = blk & 15;
  const int hy = (blk >> 4) & 15;
  const int bb = blk >> 8;

  // ---------- Phase A: LayerNorm -> xn (f16, row-major, padded stride) ----------
  {
    const int c0 = lane * 8;
    const v4f lw0 = *(const v4f*)(ln_w + c0);
    const v4f lw1 = *(const v4f*)(ln_w + c0 + 4);
    const v4f lb0 = *(const v4f*)(ln_b + c0);
    const v4f lb1 = *(const v4f*)(ln_b + c0 + 4);
#pragma unroll
    for (int tok = 0; tok < 8; ++tok) {
      const int t = wid * 8 + tok;               // token in window
      const int gy = t >> 3, gx = t & 7;
      const float* row =
          x + (((size_t)bb * Hh + (gy * 16 + hy)) * Ww + (gx * 16 + wx)) * Cc;
      v4f a0 = *(const v4f*)(row + c0);
      v4f a1 = *(const v4f*)(row + c0 + 4);
      float s = 0.f, s2 = 0.f;
#pragma unroll
      for (int i = 0; i < 4; ++i) {
        s += a0[i] + a1[i];
        s2 += a0[i] * a0[i] + a1[i] * a1[i];
      }
#pragma unroll
      for (int m = 1; m < 32; m <<= 1) {
        s += __shfl_xor(s, m);
        s2 += __shfl_xor(s2, m);
      }
      const float mean = s * (1.f / 256.f);
      const float var = s2 * (1.f / 256.f) - mean * mean;
      const float rstd = __frsqrt_rn(var + 1e-5f);
      v8h o;
#pragma unroll
      for (int i = 0; i < 4; ++i) {
        o[i]     = (f16)(((a0[i] - mean) * rstd) * lw0[i] + lb0[i]);
        o[i + 4] = (f16)(((a1[i] - mean) * rstd) * lw1[i] + lb1[i]);
      }
      *(v8h*)(xn + t * XN_STRIDE + c0) = o;
    }
  }
  __syncthreads();

  // ---------- Phase B: QKV GEMM (M=64, N=768, K=256) ----------
  // 3 region passes; each wave does 2 n-tiles per region (6 tiles total).
  qkv_region<0>(xn, qs, kf, vf, wqkv_frag, in_proj_b, wid, lane, lidlow, hi, klo);
  qkv_region<1>(xn, qs, kf, vf, wqkv_frag, in_proj_b, wid, lane, lidlow, hi, klo);
  qkv_region<2>(xn, qs, kf, vf, wqkv_frag, in_proj_b, wid, lane, lidlow, hi, klo);
  __syncthreads();

  // ---------- Phase C: attention, wave == head (no cross-wave deps) ----------
  {
    const int hd = wid;
    f16* paw = pa + hd * 64 * 64;
    const f16* qh = qs + hd * 64 * Q_STRIDE;
    const f16* kfh = kf + hd * 4 * 32 * 16;
    const f16* vfh = vf + hd * 2 * 2 * 32 * 16;

#pragma unroll 1
    for (int mt = 0; mt < 4; ++mt) {
      const v16h aq = load_a_lds(qh + (mt * 16 + lidlow) * Q_STRIDE, klo);
      v8f sc[4];
#pragma unroll
      for (int nt2 = 0; nt2 < 4; ++nt2) {
        const v16h bk = *(const v16h*)(kfh + (nt2 * 32 + lane) * 16);
        v8f z = {};
        sc[nt2] = wmma16(aq, bk, z);             // K=32 == DH: single WMMA
      }
      // softmax over 64 columns; row data lives in a 16-lane half (xor<=8 stays inside)
#pragma unroll
      for (int r = 0; r < 8; ++r) {
        float mx = fmaxf(fmaxf(sc[0][r], sc[1][r]), fmaxf(sc[2][r], sc[3][r]));
#pragma unroll
        for (int m2 = 1; m2 < 16; m2 <<= 1) mx = fmaxf(mx, __shfl_xor(mx, m2));
        float e0 = __expf(sc[0][r] - mx), e1 = __expf(sc[1][r] - mx);
        float e2 = __expf(sc[2][r] - mx), e3 = __expf(sc[3][r] - mx);
        float sum = e0 + e1 + e2 + e3;
#pragma unroll
        for (int m2 = 1; m2 < 16; m2 <<= 1) sum += __shfl_xor(sum, m2);
        const float inv = __frcp_rn(sum);
        const int l = mt * 16 + r + 8 * hi;
        paw[l * 64 + 0 * 16 + lidlow] = (f16)(e0 * inv);
        paw[l * 64 + 1 * 16 + lidlow] = (f16)(e1 * inv);
        paw[l * 64 + 2 * 16 + lidlow] = (f16)(e2 * inv);
        paw[l * 64 + 3 * 16 + lidlow] = (f16)(e3 * inv);
      }
    }

    // out = attn @ V (M=64, N=32, K=64); attn-out overlays xn (dead after phase B)
#pragma unroll 1
    for (int mt = 0; mt < 4; ++mt) {
      const v16h a0 = load_a_lds(paw + (mt * 16 + lidlow) * 64 + 0, klo);
      const v16h a1 = load_a_lds(paw + (mt * 16 + lidlow) * 64 + 32, klo);
#pragma unroll
      for (int dt = 0; dt < 2; ++dt) {
        v8f acc = {};
        const v16h b0 = *(const v16h*)(vfh + ((dt * 2 + 0) * 32 + lane) * 16);
        const v16h b1 = *(const v16h*)(vfh + ((dt * 2 + 1) * 32 + lane) * 16);
        acc = wmma16(a0, b0, acc);
        acc = wmma16(a1, b1, acc);
        const int n = hd * 32 + dt * 16 + lidlow;
#pragma unroll
        for (int r = 0; r < 8; ++r) {
          const int m = mt * 16 + r + 8 * hi;
          xn[m * XN_STRIDE + n] = (f16)acc[r];
        }
      }
    }
  }
  __syncthreads();

  // ---------- Phase D: out-proj GEMM + bias + LayerScale + grid reverse ----------
  {
#pragma unroll 1
    for (int j = 0; j < 2; ++j) {
      const int ntg = wid * 2 + j;
      v8f acc[4] = {};
#pragma unroll 1
      for (int ks = 0; ks < 8; ++ks) {
        const v16h bfr =
            *(const v16h*)(wout_frag + (((size_t)ntg * 8 + ks) * 32 + lane) * 16);
#pragma unroll
        for (int mt = 0; mt < 4; ++mt) {
          const v16h afr =
              load_a_lds(xn + (mt * 16 + lidlow) * XN_STRIDE + ks * 32, klo);
          acc[mt] = wmma16(afr, bfr, acc[mt]);
        }
      }
      const int n = ntg * 16 + lidlow;
      const float bias = out_proj_b[n];
      const float g = gamma[n];
#pragma unroll
      for (int mt = 0; mt < 4; ++mt) {
#pragma unroll
        for (int r = 0; r < 8; ++r) {
          const int m = mt * 16 + r + 8 * hi;
          const int gy = m >> 3, gx = m & 7;
          const float v = (acc[mt][r] + bias) * g;
          out[(((size_t)bb * Hh + (gy * 16 + hy)) * Ww + (gx * 16 + wx)) * Cc + n] = v;
        }
      }
    }
  }
}

extern "C" void kernel_launch(void* const* d_in, const int* in_sizes, int n_in,
                              void* d_out, int out_size, void* d_ws, size_t ws_size,
                              hipStream_t stream) {
  const float* x          = (const float*)d_in[0];
  const float* ln_w       = (const float*)d_in[1];
  const float* ln_b       = (const float*)d_in[2];
  const float* in_proj_w  = (const float*)d_in[3];
  const float* in_proj_b  = (const float*)d_in[4];
  const float* out_proj_w = (const float*)d_in[5];
  const float* out_proj_b = (const float*)d_in[6];
  const float* gamma      = (const float*)d_in[7];

  f16* wqkv_frag = (f16*)d_ws;
  f16* wout_frag = wqkv_frag + WQKV_HALF_COUNT;

  hipFuncSetAttribute(reinterpret_cast<const void*>(&grid_attn_kernel),
                      hipFuncAttributeMaxDynamicSharedMemorySize, SMEM_BYTES);

  const int total = WQKV_HALF_COUNT + WOUT_HALF_COUNT;
  pack_weights_kernel<<<(total + 255) / 256, 256, 0, stream>>>(
      in_proj_w, out_proj_w, wqkv_frag, wout_frag);

  grid_attn_kernel<<<4096, TPB, SMEM_BYTES, stream>>>(
      x, ln_w, ln_b, wqkv_frag, in_proj_b, wout_frag, out_proj_b, gamma,
      (float*)d_out);
}
